// _PointnetSAModuleBase_39539468927434
// MI455X (gfx1250) — matrix operand order repaired and checked by reading, not compiled, not run on hardware
//
#include <hip/hip_runtime.h>
#include <hip/hip_bf16.h>

// ---------------------------------------------------------------------------
// PointNet++ SA module for MI455X (gfx1250, wave32).
//   B=8, N=8192, NPOINT=2048, NSAMPLE=32, C=64, MLP 67->64->64->128, r=0.2
// Outputs (concatenated in d_out):
//   new_xyz      : (B, NPOINT, 3)    = 49152 floats
//   new_features : (B, 128, NPOINT)  = 2097152 floats
// ---------------------------------------------------------------------------

#define Bsz     8
#define Npts    8192
#define NPOINT  2048
#define NSAMPLE 32
#define CIN     64
#define R2      0.04f

typedef __attribute__((ext_vector_type(16))) _Float16 v16h;
typedef __attribute__((ext_vector_type(8)))  float    v8f;

// ---------------------------------------------------------------------------
// Kernel 1: furthest point sampling. One 1024-thread block per batch.
// Each thread owns 8 points in registers (coords + running min-dist).
// ---------------------------------------------------------------------------
__global__ __launch_bounds__(1024) void fps_kernel(const float* __restrict__ xyz,
                                                   float* __restrict__ new_xyz) {
    const int b   = blockIdx.x;
    const int tid = threadIdx.x;
    const int lane = tid & 31;
    const int wave = tid >> 5;

    __shared__ float rd[32], rx[32], ry[32], rz[32];
    __shared__ int   rj[32];
    __shared__ float scx, scy, scz;

    const float* base = xyz + (size_t)b * Npts * 3;

    float px[8], py[8], pz[8], pd[8];
#pragma unroll
    for (int q = 0; q < 8; ++q) {
        int i = q * 1024 + tid;
        px[q] = base[i * 3 + 0];
        py[q] = base[i * 3 + 1];
        pz[q] = base[i * 3 + 2];
        pd[q] = 3.4e38f;
    }
    if (tid == 0) { scx = base[0]; scy = base[1]; scz = base[2]; }
    __syncthreads();

    for (int it = 0; it < NPOINT; ++it) {
        const float cx = scx, cy = scy, cz = scz;
        if (tid == 0) {
            float* o = new_xyz + ((size_t)b * NPOINT + it) * 3;
            o[0] = cx; o[1] = cy; o[2] = cz;
        }
        float bd = -1.0f, bx = 0.f, by = 0.f, bz = 0.f;
        int   bj = 0;
#pragma unroll
        for (int q = 0; q < 8; ++q) {
            float dx = px[q] - cx, dy = py[q] - cy, dz = pz[q] - cz;
            float d  = dx * dx + dy * dy + dz * dz;
            d = fminf(d, pd[q]);
            pd[q] = d;
            if (d > bd) { bd = d; bj = q * 1024 + tid; bx = px[q]; by = py[q]; bz = pz[q]; }
        }
        // wave32 argmax reduction
#pragma unroll
        for (int off = 16; off > 0; off >>= 1) {
            float od = __shfl_down(bd, off, 32);
            int   oj = __shfl_down(bj, off, 32);
            float ox = __shfl_down(bx, off, 32);
            float oy = __shfl_down(by, off, 32);
            float oz = __shfl_down(bz, off, 32);
            if (od > bd) { bd = od; bj = oj; bx = ox; by = oy; bz = oz; }
        }
        if (lane == 0) { rd[wave] = bd; rj[wave] = bj; rx[wave] = bx; ry[wave] = by; rz[wave] = bz; }
        __syncthreads();
        if (wave == 0) {
            bd = rd[lane]; bj = rj[lane]; bx = rx[lane]; by = ry[lane]; bz = rz[lane];
#pragma unroll
            for (int off = 16; off > 0; off >>= 1) {
                float od = __shfl_down(bd, off, 32);
                int   oj = __shfl_down(bj, off, 32);
                float ox = __shfl_down(bx, off, 32);
                float oy = __shfl_down(by, off, 32);
                float oz = __shfl_down(bz, off, 32);
                if (od > bd) { bd = od; bj = oj; bx = ox; by = oy; bz = oz; }
            }
            if (lane == 0) { scx = bx; scy = by; scz = bz; }
        }
        __syncthreads();
    }
}

// ---------------------------------------------------------------------------
// Kernel 2: ball query. One wave32 per query center; ordered first-32 gather
// via ballot + prefix popcount.
// ---------------------------------------------------------------------------
#define BQ_WAVES 8
__global__ __launch_bounds__(256) void ballquery_kernel(const float* __restrict__ xyz,
                                                        const float* __restrict__ new_xyz,
                                                        int* __restrict__ idx_out) {
    __shared__ int sIdx[BQ_WAVES][NSAMPLE];
    const int wave = threadIdx.x >> 5;
    const int lane = threadIdx.x & 31;
    const int s  = blockIdx.x * BQ_WAVES + wave;   // 0 .. B*NPOINT-1
    const int b  = s / NPOINT;

    const float* c = new_xyz + (size_t)s * 3;
    const float cx = c[0], cy = c[1], cz = c[2];
    const float* P = xyz + (size_t)b * Npts * 3;

    int cnt = 0;
    for (int basej = 0; basej < Npts && cnt < NSAMPLE; basej += 32) {
        const int j = basej + lane;
        float dx = P[j * 3 + 0] - cx;
        float dy = P[j * 3 + 1] - cy;
        float dz = P[j * 3 + 2] - cz;
        const bool inb = (dx * dx + dy * dy + dz * dz) < R2;
        unsigned long long mask = __ballot(inb);
        if (inb) {
            int pos = cnt + __popcll(mask & ((1ull << lane) - 1ull));
            if (pos < NSAMPLE) sIdx[wave][pos] = j;
        }
        cnt += __popcll(mask);
    }
    __syncthreads();
    const int total = cnt < NSAMPLE ? cnt : NSAMPLE;
    int first = 0;
    if (total > 0) first = sIdx[wave][0];
    const int v = (lane < total) ? sIdx[wave][lane] : first;
    idx_out[(size_t)s * NSAMPLE + lane] = v;
}

// ---------------------------------------------------------------------------
// Kernel 3: gather + 3-layer MLP (WMMA f16->f32) + max pool over 32 samples.
// One wave32 per sampled point; 2 waves per block; each wave loops over
// POINTS_PER_WAVE points so LDS weight staging is amortized 8x.
//
// Weights staged in LDS in native (cout, cin) f16 layout: a B fragment is
// one contiguous, 32B-aligned v16h per lane (lane l holds column n = l%16,
// halves h -> K = kbase + 16*(l/16) + h).
//
// Activations staged in natural K order within each 32-wide K-chunk (swz is identity),
// so an A fragment is also one
// contiguous, 32B-aligned v16h per lane at row*ld + chunk + 16*(l/16).
// ---------------------------------------------------------------------------
#define MLP_WAVES 2
#define PPW 8        // points processed per wave
#define LDX 96       // padded K for layer 1 (67 -> 96)

__device__ __forceinline__ int swz(int k) {   // identity: natural K order
    return k;
}

__device__ __forceinline__ v8f wmma16(v16h a, v16h b, v8f c) {
    return __builtin_amdgcn_wmma_f32_16x16x32_f16(false, a, false, b, (short)0, c,
                                                  false, false);
}

__device__ __forceinline__ v16h load_a(const _Float16* __restrict__ tile,
                                       int lane, int kchunk, int ld) {
    return *(const v16h*)(tile + (lane & 15) * ld + kchunk + 16 * (lane >> 4));
}

__device__ __forceinline__ v16h load_b(const _Float16* __restrict__ W,
                                       int lane, int kbase, int n0, int ldk) {
    return *(const v16h*)(W + (size_t)(n0 + (lane & 15)) * ldk + kbase +
                          16 * (lane >> 4));
}

__global__ __launch_bounds__(MLP_WAVES * 32) void mlp_kernel(
        const float* __restrict__ xyz, const float* __restrict__ feats,
        const int* __restrict__ idx,
        const float* __restrict__ W1, const float* __restrict__ b1,
        const float* __restrict__ W2, const float* __restrict__ b2,
        const float* __restrict__ W3, const float* __restrict__ b3,
        const float* __restrict__ new_xyz, float* __restrict__ out_feats) {

    __shared__ __attribute__((aligned(32))) _Float16 Wh1[64 * 96];    // [n][k]
    __shared__ __attribute__((aligned(32))) _Float16 Wh2[64 * 64];    // [n][k]
    __shared__ __attribute__((aligned(32))) _Float16 Wh3[128 * 64];   // [n][k]
    __shared__ __attribute__((aligned(32))) _Float16 Xb[MLP_WAVES][32 * LDX];
    __shared__ __attribute__((aligned(32))) _Float16 Hb[MLP_WAVES][32 * 64];

    const int tid  = threadIdx.x;
    const int wave = tid >> 5;
    const int lane = tid & 31;

    // ---- stage weights once per block: native (n,k) f16, K-padded for L1 ----
    for (int i = tid; i < 64 * 96; i += MLP_WAVES * 32) {
        int n = i / 96, k = i % 96;
        Wh1[i] = (k < 67) ? (_Float16)W1[n * 67 + k] : (_Float16)0.0f;
    }
    for (int i = tid; i < 64 * 64; i += MLP_WAVES * 32)
        Wh2[i] = (_Float16)W2[i];
    for (int i = tid; i < 128 * 64; i += MLP_WAVES * 32)
        Wh3[i] = (_Float16)W3[i];
    __syncthreads();

    _Float16* X = &Xb[wave][0];
    _Float16* H = &Hb[wave][0];

#pragma unroll 1
    for (int pt = 0; pt < PPW; ++pt) {
        const int s  = (blockIdx.x * MLP_WAVES + wave) * PPW + pt; // 0..B*NPOINT-1
        const int b  = s / NPOINT;
        const int sp = s % NPOINT;

        // ---- gather: lane = sample row; 3 centered coords + 64 features ----
        {
            const int r = lane;
            const int j = idx[((size_t)s) * NSAMPLE + r];
            const float* cc = new_xyz + (size_t)s * 3;
            const float* p  = xyz + ((size_t)b * Npts + j) * 3;
            _Float16* xr = X + r * LDX;
            // zero the tail chunk (positions 64..95), then fill channels 64..66
#pragma unroll
            for (int k = 64; k < LDX; ++k) xr[k] = (_Float16)0.0f;
            xr[swz(0)] = (_Float16)(p[0] - cc[0]);
            xr[swz(1)] = (_Float16)(p[1] - cc[1]);
            xr[swz(2)] = (_Float16)(p[2] - cc[2]);
            const float* fb = feats + (size_t)b * CIN * Npts + j;
#pragma unroll 8
            for (int c = 0; c < CIN; ++c)
                xr[(3 + c) >= 64 ? (3 + c) : swz(3 + c)] =
                    (_Float16)fb[(size_t)c * Npts];
            // channels 64..66 land in chunk 2 where swz == identity
        }
        __syncthreads();

        // ---- layer 1: (32x96) x (96x64), relu -> H (natural K order) ----
#pragma unroll
        for (int mt = 0; mt < 2; ++mt) {
            const v16h a0 = load_a(X + mt * 16 * LDX, lane, 0,  LDX);
            const v16h a1 = load_a(X + mt * 16 * LDX, lane, 32, LDX);
            const v16h a2 = load_a(X + mt * 16 * LDX, lane, 64, LDX);
#pragma unroll
            for (int nt = 0; nt < 4; ++nt) {
                v8f acc = {};
                acc = wmma16(a0, load_b(Wh1, lane, 0,  nt * 16, 96), acc);
                acc = wmma16(a1, load_b(Wh1, lane, 32, nt * 16, 96), acc);
                acc = wmma16(a2, load_b(Wh1, lane, 64, nt * 16, 96), acc);
                const int col  = nt * 16 + (lane & 15);
                const int scol = swz(col);
                const int g    = lane >> 4;
                const float bias = b1[col];
#pragma unroll
                for (int v = 0; v < 8; ++v) {
                    float h = fmaxf(acc[v] + bias, 0.0f);
                    H[(mt * 16 + v + 8 * g) * 64 + scol] = (_Float16)h;
                }
            }
        }
        __syncthreads();

        // ---- layer 2: (32x64) x (64x64), relu -> X (reused, ld=LDX) ----
#pragma unroll
        for (int mt = 0; mt < 2; ++mt) {
            const v16h a0 = load_a(H + mt * 16 * 64, lane, 0,  64);
            const v16h a1 = load_a(H + mt * 16 * 64, lane, 32, 64);
#pragma unroll
            for (int nt = 0; nt < 4; ++nt) {
                v8f acc = {};
                acc = wmma16(a0, load_b(Wh2, lane, 0,  nt * 16, 64), acc);
                acc = wmma16(a1, load_b(Wh2, lane, 32, nt * 16, 64), acc);
                const int col  = nt * 16 + (lane & 15);
                const int scol = swz(col);
                const int g    = lane >> 4;
                const float bias = b2[col];
#pragma unroll
                for (int v = 0; v < 8; ++v) {
                    float h = fmaxf(acc[v] + bias, 0.0f);
                    X[(mt * 16 + v + 8 * g) * LDX + scol] = (_Float16)h;
                }
            }
        }
        __syncthreads();

        // ---- layer 3: (32x64) x (64x128), relu, column-max over 32 rows ----
        const v16h A00 = load_a(X, lane, 0,  LDX);
        const v16h A01 = load_a(X, lane, 32, LDX);
        const v16h A10 = load_a(X + 16 * LDX, lane, 0,  LDX);
        const v16h A11 = load_a(X + 16 * LDX, lane, 32, LDX);
#pragma unroll
        for (int nt = 0; nt < 8; ++nt) {
            const float bias = b3[nt * 16 + (lane & 15)];
            float cmax = 0.0f;   // relu output is >= 0
#pragma unroll
            for (int mt = 0; mt < 2; ++mt) {
                v8f acc = {};
                acc = wmma16(mt ? A10 : A00, load_b(Wh3, lane, 0,  nt * 16, 64), acc);
                acc = wmma16(mt ? A11 : A01, load_b(Wh3, lane, 32, nt * 16, 64), acc);
#pragma unroll
                for (int v = 0; v < 8; ++v)
                    cmax = fmaxf(cmax, fmaxf(acc[v] + bias, 0.0f));
            }
            // lanes l and l^16 cover the same column (other 16 rows)
            cmax = fmaxf(cmax, __shfl_xor(cmax, 16, 32));
            if (lane < 16) {
                const int ch = nt * 16 + lane;
                out_feats[((size_t)b * 128 + ch) * NPOINT + sp] = cmax;
            }
        }
        __syncthreads();
    }
}

// ---------------------------------------------------------------------------
extern "C" void kernel_launch(void* const* d_in, const int* in_sizes, int n_in,
                              void* d_out, int out_size, void* d_ws, size_t ws_size,
                              hipStream_t stream) {
    const float* xyz   = (const float*)d_in[0];
    const float* feats = (const float*)d_in[1];
    const float* W1 = (const float*)d_in[2];
    const float* b1 = (const float*)d_in[3];
    const float* W2 = (const float*)d_in[4];
    const float* b2 = (const float*)d_in[5];
    const float* W3 = (const float*)d_in[6];
    const float* b3 = (const float*)d_in[7];

    float* new_xyz   = (float*)d_out;
    float* out_feats = (float*)d_out + (size_t)Bsz * NPOINT * 3;
    int*   ball_idx  = (int*)d_ws;   // B*NPOINT*NSAMPLE ints = 2 MiB

    fps_kernel<<<Bsz, 1024, 0, stream>>>(xyz, new_xyz);

    ballquery_kernel<<<(Bsz * NPOINT) / BQ_WAVES, BQ_WAVES * 32, 0, stream>>>(
        xyz, new_xyz, ball_idx);

    mlp_kernel<<<(Bsz * NPOINT) / (MLP_WAVES * PPW), MLP_WAVES * 32, 0, stream>>>(
        xyz, feats, ball_idx, W1, b1, W2, b2, W3, b3, new_xyz, out_feats);
}
